// QwenTextEncoder_39934605918632
// MI455X (gfx1250) — matrix operand order repaired and checked by reading, not compiled, not run on hardware
//
#include <hip/hip_runtime.h>

// ---------------- model constants ----------------
#define LAYERS 2
#define HDIM   2048
#define NHEADS 16
#define NKVH   2
#define HEADD  128
#define FFDIM  11008
#define BSZ    32
#define SEQ    64
#define MT     (BSZ*SEQ)   // 2048 tokens
#define LR     16
#define PHD    1024
#define TDD    512

// ---------------- WMMA types ----------------
typedef __attribute__((ext_vector_type(16))) __bf16 v16bf;
typedef __attribute__((ext_vector_type(8)))  __bf16 v8bf;
typedef __attribute__((ext_vector_type(8)))  float  v8f;
typedef __attribute__((ext_vector_type(4)))  float  v4f;

#define LDST 72   // ushort stride per 64-element LDS row (+8 pad: 16B-aligned chunks, bank spread)

__device__ __forceinline__ unsigned short f2bf(float f) {
    union { float f; unsigned u; } v; v.f = f;
    unsigned u = v.u;
    unsigned r = u + 0x7FFFu + ((u >> 16) & 1u);
    return (unsigned short)(r >> 16);
}

__device__ __forceinline__ v16bf ldfrag(const unsigned short* p0, const unsigned short* p1) {
    v8bf a = *(const v8bf*)p0;
    v8bf b = *(const v8bf*)p1;
    return __builtin_shufflevector(a, b, 0,1,2,3,4,5,6,7,8,9,10,11,12,13,14,15);
}

// stage 16 fp32 -> bf16 into LDS (one thread handles 16 contiguous elements)
__device__ __forceinline__ void stage16(unsigned short* dst, const float* src, bool valid) {
    if (valid) {
#pragma unroll
        for (int t = 0; t < 4; ++t) {
            v4f x = *(const v4f*)(src + t*4);
            dst[t*4+0] = f2bf(x[0]); dst[t*4+1] = f2bf(x[1]);
            dst[t*4+2] = f2bf(x[2]); dst[t*4+3] = f2bf(x[3]);
        }
    } else {
#pragma unroll
        for (int t = 0; t < 16; ++t) dst[t] = 0;
    }
}

// ---------------- generic GEMM: C = A(MxK) * W(NxK)^T [+bias][+addsrc] ----------------
// block tile 128x128, 8 waves of 32 lanes, each wave computes 32x64 (2x4 wmma tiles)
// K staged in chunks of 64 (two 32-K WMMA steps per barrier pair)
__global__ __launch_bounds__(256) void gemm_wmma_kernel(
    const float* __restrict__ A, const float* __restrict__ W,
    const float* __restrict__ bias, const float* __restrict__ addsrc,
    float* __restrict__ C, int M, int N, int K)
{
    __shared__ unsigned short sA[128*LDST];
    __shared__ unsigned short sB[128*LDST];
    const int tid  = threadIdx.x;
    const int lane = tid & 31;
    const int wave = tid >> 5;
    const int wm   = wave & 3;
    const int wn   = wave >> 2;
    const int mBlk = blockIdx.y * 128;
    const int nBlk = blockIdx.x * 128;

    v8f acc[2][4];
#pragma unroll
    for (int i = 0; i < 2; ++i)
#pragma unroll
        for (int j = 0; j < 4; ++j)
#pragma unroll
            for (int e = 0; e < 8; ++e) acc[i][j][e] = 0.f;

    const int srow = tid >> 2;          // 0..63
    const int scol = (tid & 3) * 16;    // 0,16,32,48
    const int half = lane >> 4;
    const int l16  = lane & 15;

    for (int k0 = 0; k0 < K; k0 += 64) {
#pragma unroll
        for (int p = 0; p < 2; ++p) {
            int row = srow + p*64;
            int grA = mBlk + row;
            int grB = nBlk + row;
            stage16(&sA[row*LDST + scol], A + (size_t)grA*K + k0 + scol, grA < M);
            stage16(&sB[row*LDST + scol], W + (size_t)grB*K + k0 + scol, grB < N);
        }
        // prefetch next K tile while this stage computes
        if (k0 + 64 < K) {
            int grA = mBlk + srow;
            int grB = nBlk + srow;
            if (grA < M) __builtin_prefetch(A + (size_t)grA*K + k0 + 64 + scol, 0, 1);
            if (grB < N) __builtin_prefetch(W + (size_t)grB*K + k0 + 64 + scol, 0, 1);
        }
        __syncthreads();

#pragma unroll
        for (int ks = 0; ks < 2; ++ks) {
            const int kof = ks * 32;
            v16bf af[2], bfr[4];
#pragma unroll
            for (int i = 0; i < 2; ++i) {
                const unsigned short* base = &sA[(wm*32 + i*16 + l16)*LDST + kof];
                af[i] = ldfrag(base + half*16, base + half*16 + 8);
            }
#pragma unroll
            for (int j = 0; j < 4; ++j) {
                const unsigned short* base = &sB[(wn*64 + j*16 + l16)*LDST + kof];
                bfr[j] = ldfrag(base + half*16, base + half*16 + 8);
            }
#pragma unroll
            for (int i = 0; i < 2; ++i)
#pragma unroll
                for (int j = 0; j < 4; ++j)
                    acc[i][j] = __builtin_amdgcn_wmma_f32_16x16x32_bf16(
                        false, af[i], false, bfr[j], (short)0, acc[i][j], false, false);
        }
        __syncthreads();
    }

#pragma unroll
    for (int i = 0; i < 2; ++i) {
#pragma unroll
        for (int j = 0; j < 4; ++j) {
            int n = nBlk + wn*64 + j*16 + l16;
            if (n >= N) continue;
            float bv = bias ? bias[n] : 0.f;
#pragma unroll
            for (int e = 0; e < 8; ++e) {
                int m = mBlk + wm*32 + i*16 + half*8 + e;
                if (m >= M) continue;
                size_t idx = (size_t)m * N + n;
                float val = acc[i][j][e] + bv;
                if (addsrc) val += addsrc[idx];
                C[idx] = val;
            }
        }
    }
}

// ---------------- fused gate/up: Out = silu(A*Wg^T) * (A*Wu^T), tile 128x64 ----------------
__global__ __launch_bounds__(256) void gateup_wmma_kernel(
    const float* __restrict__ A, const float* __restrict__ Wg,
    const float* __restrict__ Wu, float* __restrict__ Out,
    int M, int N, int K)
{
    __shared__ unsigned short sA[128*LDST];
    __shared__ unsigned short sG[64*LDST];
    __shared__ unsigned short sU[64*LDST];
    const int tid  = threadIdx.x;
    const int lane = tid & 31;
    const int wave = tid >> 5;
    const int wm   = wave & 3;     // rows: 32 each
    const int wn   = wave >> 2;    // cols: 32 each (block N=64)
    const int mBlk = blockIdx.y * 128;
    const int nBlk = blockIdx.x * 64;
    const int half = lane >> 4;
    const int l16  = lane & 15;

    v8f aG[2][2], aU[2][2];
#pragma unroll
    for (int i = 0; i < 2; ++i)
#pragma unroll
        for (int j = 0; j < 2; ++j)
#pragma unroll
            for (int e = 0; e < 8; ++e) { aG[i][j][e] = 0.f; aU[i][j][e] = 0.f; }

    const int srow = tid >> 2;          // A: 0..63
    const int scol = (tid & 3) * 16;
    const int bsel = tid >> 7;          // 0: gate, 1: up
    const int t2   = tid & 127;
    const int brow = t2 >> 2;           // 0..31
    const int bcol = (t2 & 3) * 16;

    for (int k0 = 0; k0 < K; k0 += 64) {
#pragma unroll
        for (int p = 0; p < 2; ++p) {
            int row = srow + p*64;
            int gr = mBlk + row;
            stage16(&sA[row*LDST + scol], A + (size_t)gr*K + k0 + scol, gr < M);
        }
        const float* Wsrc = bsel ? Wu : Wg;
        unsigned short* sW = bsel ? sU : sG;
#pragma unroll
        for (int p = 0; p < 2; ++p) {
            int row = brow + p*32;
            int gr = nBlk + row;
            stage16(&sW[row*LDST + bcol], Wsrc + (size_t)gr*K + k0 + bcol, gr < N);
        }
        if (k0 + 64 < K) {
            int grA = mBlk + srow;
            int grW = nBlk + brow;
            if (grA < M) __builtin_prefetch(A + (size_t)grA*K + k0 + 64 + scol, 0, 1);
            if (grW < N) __builtin_prefetch(Wsrc + (size_t)grW*K + k0 + 64 + bcol, 0, 1);
        }
        __syncthreads();

#pragma unroll
        for (int ks = 0; ks < 2; ++ks) {
            const int kof = ks * 32;
            v16bf af[2], bg[2], bu[2];
#pragma unroll
            for (int i = 0; i < 2; ++i) {
                const unsigned short* base = &sA[(wm*32 + i*16 + l16)*LDST + kof];
                af[i] = ldfrag(base + half*16, base + half*16 + 8);
            }
#pragma unroll
            for (int j = 0; j < 2; ++j) {
                const unsigned short* bg_ = &sG[(wn*32 + j*16 + l16)*LDST + kof];
                const unsigned short* bu_ = &sU[(wn*32 + j*16 + l16)*LDST + kof];
                bg[j] = ldfrag(bg_ + half*16, bg_ + half*16 + 8);
                bu[j] = ldfrag(bu_ + half*16, bu_ + half*16 + 8);
            }
#pragma unroll
            for (int i = 0; i < 2; ++i)
#pragma unroll
                for (int j = 0; j < 2; ++j) {
                    aG[i][j] = __builtin_amdgcn_wmma_f32_16x16x32_bf16(
                        false, af[i], false, bg[j], (short)0, aG[i][j], false, false);
                    aU[i][j] = __builtin_amdgcn_wmma_f32_16x16x32_bf16(
                        false, af[i], false, bu[j], (short)0, aU[i][j], false, false);
                }
        }
        __syncthreads();
    }

#pragma unroll
    for (int i = 0; i < 2; ++i) {
#pragma unroll
        for (int j = 0; j < 2; ++j) {
            int n = nBlk + wn*32 + j*16 + l16;
            if (n >= N) continue;
#pragma unroll
            for (int e = 0; e < 8; ++e) {
                int m = mBlk + wm*32 + i*16 + half*8 + e;
                if (m >= M) continue;
                float g = aG[i][j][e];
                float u = aU[i][j][e];
                float s = g / (1.f + __expf(-g));   // silu
                Out[(size_t)m * N + n] = s * u;
            }
        }
    }
}

// ---------------- embedding gather ----------------
__global__ __launch_bounds__(256) void embed_kernel(
    const int* __restrict__ ids, const float* __restrict__ E, float* __restrict__ Hh)
{
    int row = blockIdx.x;
    int id  = ids[row];
    const float* src = E + (size_t)id * HDIM;
    float* dst = Hh + (size_t)row * HDIM;
    for (int d = threadIdx.x; d < HDIM; d += 256) dst[d] = src[d];
}

// ---------------- RMSNorm ----------------
__global__ __launch_bounds__(256) void rmsnorm_kernel(
    const float* __restrict__ X, const float* __restrict__ w,
    float* __restrict__ Y, int Hd)
{
    int row = blockIdx.x;
    const float* x = X + (size_t)row * Hd;
    float ss = 0.f;
    for (int i = threadIdx.x; i < Hd; i += 256) { float v = x[i]; ss += v*v; }
    for (int off = 16; off; off >>= 1) ss += __shfl_xor(ss, off, 32);
    __shared__ float red[8];
    __shared__ float rtot;
    if ((threadIdx.x & 31) == 0) red[threadIdx.x >> 5] = ss;
    __syncthreads();
    if (threadIdx.x == 0) {
        float t = 0.f;
        for (int i = 0; i < 8; ++i) t += red[i];
        rtot = rsqrtf(t / (float)Hd + 1e-6f);
    }
    __syncthreads();
    float r = rtot;
    for (int i = threadIdx.x; i < Hd; i += 256)
        Y[(size_t)row * Hd + i] = x[i] * r * w[i];
}

// ---------------- LoRA down: T[m,r] = sum_k X[m,k]*Aw[r,k] ----------------
__global__ __launch_bounds__(256) void lora_down_kernel(
    const float* __restrict__ X, const float* __restrict__ Aw,
    float* __restrict__ T, int K)
{
    int m = blockIdx.x;
    const float* x = X + (size_t)m * K;
    __shared__ float red[8];
    for (int r = 0; r < LR; ++r) {
        const float* a = Aw + (size_t)r * K;
        float s = 0.f;
        for (int i = threadIdx.x; i < K; i += 256) s += x[i] * a[i];
        for (int off = 16; off; off >>= 1) s += __shfl_xor(s, off, 32);
        if ((threadIdx.x & 31) == 0) red[threadIdx.x >> 5] = s;
        __syncthreads();
        if (threadIdx.x == 0) {
            float t = 0.f;
            for (int i = 0; i < 8; ++i) t += red[i];
            T[m*LR + r] = t;
        }
        __syncthreads();
    }
}

// ---------------- LoRA up add: Y[m,n] += 2 * sum_r T[m,r]*Bw[n,r] ----------------
__global__ __launch_bounds__(256) void lora_up_kernel(
    const float* __restrict__ T, const float* __restrict__ Bw,
    float* __restrict__ Y, int M, int N)
{
    int idx = blockIdx.x * 256 + threadIdx.x;
    if (idx >= M*N) return;
    int m = idx / N, n = idx % N;
    const float* t = T + m*LR;
    const float* b = Bw + n*LR;
    float s = 0.f;
#pragma unroll
    for (int r = 0; r < LR; ++r) s += t[r] * b[r];
    Y[idx] += 2.0f * s;
}

// ---------------- RoPE (in place): layout [tok, nh*128] ----------------
__global__ __launch_bounds__(256) void rope_kernel(float* __restrict__ Q, int nh, int total)
{
    int idx = blockIdx.x * 256 + threadIdx.x;
    if (idx >= total) return;
    int i   = idx & 63;
    int hh  = (idx >> 6) % nh;
    int tok = idx / (64 * nh);
    int s   = tok % SEQ;
    float ang = (float)s * __powf(1.0e6f, -(float)i / 64.0f);
    float c = __cosf(ang), sn = __sinf(ang);
    size_t base = (size_t)tok * (nh*HEADD) + hh*HEADD;
    float x0 = Q[base + i], x1 = Q[base + i + 64];
    Q[base + i]      = x0*c - x1*sn;
    Q[base + i + 64] = x1*c + x0*sn;
}

// ---------------- fused attention: one block per (head, batch) ----------------
#define SWZ(j,d) ((j)*HEADD + (((d)+(j)) & 127))
__global__ __launch_bounds__(256) void attn_kernel(
    const float* __restrict__ Q, const float* __restrict__ Kb,
    const float* __restrict__ Vb, const int* __restrict__ mask,
    float* __restrict__ O)
{
    int h  = blockIdx.x;
    int b  = blockIdx.y;
    int kh = h >> 3;   // GQA: 16 heads -> 2 kv heads
    __shared__ float ks[SEQ*HEADD];
    __shared__ float vs[SEQ*HEADD];
    int tid = threadIdx.x;
    for (int idx = tid; idx < SEQ*HEADD; idx += 256) {
        int j = idx >> 7, d = idx & 127;
        size_t kbase = (size_t)(b*SEQ + j) * (NKVH*HEADD) + kh*HEADD + d;
        ks[SWZ(j,d)] = Kb[kbase];
        vs[SWZ(j,d)] = Vb[kbase];
    }
    __syncthreads();
    int wave = tid >> 5, lane = tid & 31;
    int mk0 = mask[b*SEQ + lane];
    int mk1 = mask[b*SEQ + lane + 32];
    const float scale = 0.08838834764831845f;  // 1/sqrt(128)
    for (int t = 0; t < 8; ++t) {
        int i = wave + 8*t;
        const float* qrow = Q + (size_t)(b*SEQ + i) * (NHEADS*HEADD) + h*HEADD;
        float s0 = 0.f, s1 = 0.f;
        for (int d = 0; d < HEADD; ++d) {
            float qv = qrow[d];
            s0 += qv * ks[SWZ(lane, d)];
            s1 += qv * ks[SWZ(lane+32, d)];
        }
        s0 = s0*scale + ((lane      <= i && mk0) ? 0.f : -1e9f);
        s1 = s1*scale + ((lane + 32 <= i && mk1) ? 0.f : -1e9f);
        float mx = fmaxf(s0, s1);
        for (int off = 16; off; off >>= 1) mx = fmaxf(mx, __shfl_xor(mx, off, 32));
        float p0 = __expf(s0 - mx), p1 = __expf(s1 - mx);
        float sum = p0 + p1;
        for (int off = 16; off; off >>= 1) sum += __shfl_xor(sum, off, 32);
        float inv = 1.f / sum;
        p0 *= inv; p1 *= inv;
        float o0 = 0.f, o1 = 0.f, o2 = 0.f, o3 = 0.f;
        for (int j = 0; j < 32; ++j) {
            float pj = __shfl(p0, j, 32);
            o0 += pj * vs[SWZ(j, lane)];
            o1 += pj * vs[SWZ(j, lane+32)];
            o2 += pj * vs[SWZ(j, lane+64)];
            o3 += pj * vs[SWZ(j, lane+96)];
        }
        for (int j = 0; j < 32; ++j) {
            float pj = __shfl(p1, j, 32);
            o0 += pj * vs[SWZ(j+32, lane)];
            o1 += pj * vs[SWZ(j+32, lane+32)];
            o2 += pj * vs[SWZ(j+32, lane+64)];
            o3 += pj * vs[SWZ(j+32, lane+96)];
        }
        size_t ob = (size_t)(b*SEQ + i) * (NHEADS*HEADD) + h*HEADD;
        O[ob + lane]      = o0;
        O[ob + lane + 32] = o1;
        O[ob + lane + 64] = o2;
        O[ob + lane + 96] = o3;
    }
}

// ---------------- last-valid-token pooling ----------------
__global__ __launch_bounds__(256) void pool_kernel(
    const float* __restrict__ Xf, const int* __restrict__ mask, float* __restrict__ P)
{
    int b = blockIdx.x;
    __shared__ int sidx;
    if (threadIdx.x == 0) {
        int c = 0;
        for (int s = 0; s < SEQ; ++s) c += mask[b*SEQ + s];
        sidx = c - 1;
    }
    __syncthreads();
    int idx = sidx;
    for (int d = threadIdx.x; d < HDIM; d += 256)
        P[(size_t)b*HDIM + d] = Xf[(size_t)(b*SEQ + idx)*HDIM + d];
}

// ---------------- exact GELU (in place) ----------------
__global__ __launch_bounds__(256) void gelu_kernel(float* __restrict__ Z, int n)
{
    int i = blockIdx.x * 256 + threadIdx.x;
    if (i < n) {
        float x = Z[i];
        Z[i] = 0.5f * x * (1.f + erff(x * 0.70710678118654752f));
    }
}

// ---------------- driver ----------------
extern "C" void kernel_launch(void* const* d_in, const int* in_sizes, int n_in,
                              void* d_out, int out_size, void* d_ws, size_t ws_size,
                              hipStream_t stream)
{
    (void)in_sizes; (void)n_in; (void)out_size; (void)ws_size;
    const int*   ids   = (const int*)  d_in[0];
    const int*   amask = (const int*)  d_in[1];
    const float* emb   = (const float*)d_in[2];
    const float* Wq    = (const float*)d_in[3];
    const float* bq    = (const float*)d_in[4];
    const float* Aq    = (const float*)d_in[5];
    const float* Bq    = (const float*)d_in[6];
    const float* Wk    = (const float*)d_in[7];
    const float* bk    = (const float*)d_in[8];
    const float* Ak    = (const float*)d_in[9];
    const float* Bk    = (const float*)d_in[10];
    const float* Wv    = (const float*)d_in[11];
    const float* bv    = (const float*)d_in[12];
    const float* Av    = (const float*)d_in[13];
    const float* Bv    = (const float*)d_in[14];
    const float* Wo    = (const float*)d_in[15];
    const float* Ao    = (const float*)d_in[16];
    const float* Bo    = (const float*)d_in[17];
    const float* ln1   = (const float*)d_in[18];
    const float* ln2   = (const float*)d_in[19];
    const float* Wg    = (const float*)d_in[20];
    const float* Wu    = (const float*)d_in[21];
    const float* Wd    = (const float*)d_in[22];
    const float* normf = (const float*)d_in[23];
    const float* W1    = (const float*)d_in[24];
    const float* b1    = (const float*)d_in[25];
    const float* W2    = (const float*)d_in[26];
    const float* b2    = (const float*)d_in[27];
    float* out = (float*)d_out;

    char* wsb = (char*)d_ws;
    size_t off = 0;
    auto alloc = [&](size_t bytes) -> float* {
        float* p = (float*)(wsb + off);
        off += (bytes + 255) & ~(size_t)255;
        return p;
    };
    float* hA     = alloc((size_t)MT*HDIM*4);
    float* hB_    = alloc((size_t)MT*HDIM*4);
    float* xn     = alloc((size_t)MT*HDIM*4);
    float* qb     = alloc((size_t)MT*HDIM*4);
    float* ab     = alloc((size_t)MT*HDIM*4);
    float* kb     = alloc((size_t)MT*NKVH*HEADD*4);
    float* vb     = alloc((size_t)MT*NKVH*HEADD*4);
    float* tb     = alloc((size_t)MT*LR*4);
    float* mb     = alloc((size_t)MT*FFDIM*4);
    float* pooled = alloc((size_t)BSZ*HDIM*4);
    float* zb     = alloc((size_t)BSZ*PHD*4);

    float* hcur = hA;
    float* halt = hB_;

    embed_kernel<<<MT, 256, 0, stream>>>(ids, emb, hcur);

    const dim3 g2048(HDIM/128, MT/128);          // (16,16)
    const dim3 g256 (256/128,  MT/128);          // (2,16)
    const dim3 gGU  (FFDIM/64, MT/128);          // (172,16)

    for (int l = 0; l < LAYERS; ++l) {
        const float* Wq_l = Wq + (size_t)l*HDIM*HDIM;
        const float* bq_l = bq + (size_t)l*HDIM;
        const float* Aq_l = Aq + (size_t)l*LR*HDIM;
        const float* Bq_l = Bq + (size_t)l*HDIM*LR;
        const float* Wk_l = Wk + (size_t)l*NKVH*HEADD*HDIM;
        const float* bk_l = bk + (size_t)l*NKVH*HEADD;
        const float* Ak_l = Ak + (size_t)l*LR*HDIM;
        const float* Bk_l = Bk + (size_t)l*NKVH*HEADD*LR;
        const float* Wv_l = Wv + (size_t)l*NKVH*HEADD*HDIM;
        const float* bv_l = bv + (size_t)l*NKVH*HEADD;
        const float* Av_l = Av + (size_t)l*LR*HDIM;
        const float* Bv_l = Bv + (size_t)l*NKVH*HEADD*LR;
        const float* Wo_l = Wo + (size_t)l*HDIM*HDIM;
        const float* Ao_l = Ao + (size_t)l*LR*HDIM;
        const float* Bo_l = Bo + (size_t)l*HDIM*LR;
        const float* Wg_l = Wg + (size_t)l*FFDIM*HDIM;
        const float* Wu_l = Wu + (size_t)l*FFDIM*HDIM;
        const float* Wd_l = Wd + (size_t)l*HDIM*FFDIM;

        rmsnorm_kernel<<<MT, 256, 0, stream>>>(hcur, ln1 + (size_t)l*HDIM, xn, HDIM);

        // Q / K / V projections (WMMA) + LoRA
        gemm_wmma_kernel<<<g2048, 256, 0, stream>>>(xn, Wq_l, bq_l, nullptr, qb, MT, HDIM, HDIM);
        lora_down_kernel<<<MT, 256, 0, stream>>>(xn, Aq_l, tb, HDIM);
        lora_up_kernel<<<(MT*HDIM + 255)/256, 256, 0, stream>>>(tb, Bq_l, qb, MT, HDIM);

        gemm_wmma_kernel<<<g256, 256, 0, stream>>>(xn, Wk_l, bk_l, nullptr, kb, MT, NKVH*HEADD, HDIM);
        lora_down_kernel<<<MT, 256, 0, stream>>>(xn, Ak_l, tb, HDIM);
        lora_up_kernel<<<(MT*NKVH*HEADD + 255)/256, 256, 0, stream>>>(tb, Bk_l, kb, MT, NKVH*HEADD);

        gemm_wmma_kernel<<<g256, 256, 0, stream>>>(xn, Wv_l, bv_l, nullptr, vb, MT, NKVH*HEADD, HDIM);
        lora_down_kernel<<<MT, 256, 0, stream>>>(xn, Av_l, tb, HDIM);
        lora_up_kernel<<<(MT*NKVH*HEADD + 255)/256, 256, 0, stream>>>(tb, Bv_l, vb, MT, NKVH*HEADD);

        // RoPE
        rope_kernel<<<(MT*NHEADS*64 + 255)/256, 256, 0, stream>>>(qb, NHEADS, MT*NHEADS*64);
        rope_kernel<<<(MT*NKVH*64 + 255)/256, 256, 0, stream>>>(kb, NKVH, MT*NKVH*64);

        // attention
        attn_kernel<<<dim3(NHEADS, BSZ), 256, 0, stream>>>(qb, kb, vb, amask, ab);

        // O projection + residual (fused addsrc), + LoRA
        gemm_wmma_kernel<<<g2048, 256, 0, stream>>>(ab, Wo_l, nullptr, hcur, halt, MT, HDIM, HDIM);
        lora_down_kernel<<<MT, 256, 0, stream>>>(ab, Ao_l, tb, HDIM);
        lora_up_kernel<<<(MT*HDIM + 255)/256, 256, 0, stream>>>(tb, Bo_l, halt, MT, HDIM);
        { float* t = hcur; hcur = halt; halt = t; }

        // MLP: norm -> fused gate/up -> down + residual
        rmsnorm_kernel<<<MT, 256, 0, stream>>>(hcur, ln2 + (size_t)l*HDIM, xn, HDIM);
        gateup_wmma_kernel<<<gGU, 256, 0, stream>>>(xn, Wg_l, Wu_l, mb, MT, FFDIM, HDIM);
        gemm_wmma_kernel<<<g2048, 256, 0, stream>>>(mb, Wd_l, nullptr, hcur, halt, MT, HDIM, FFDIM);
        { float* t = hcur; hcur = halt; halt = t; }
    }

    // final norm, pool, pooled head
    rmsnorm_kernel<<<MT, 256, 0, stream>>>(hcur, normf, xn, HDIM);
    pool_kernel<<<BSZ, 256, 0, stream>>>(xn, amask, pooled);
    gemm_wmma_kernel<<<dim3(PHD/128, 1), 256, 0, stream>>>(pooled, W1, b1, nullptr, zb, BSZ, PHD, HDIM);
    gelu_kernel<<<(BSZ*PHD + 255)/256, 256, 0, stream>>>(zb, BSZ*PHD);
    gemm_wmma_kernel<<<dim3(TDD/128, 1), 256, 0, stream>>>(zb, W2, b2, nullptr, out, BSZ, TDD, PHD);
}
